// SlidingWindowAttention_6854767804603
// MI455X (gfx1250) — matrix operand order repaired and checked by reading, not compiled, not run on hardware
//
#include <hip/hip_runtime.h>
#include <hip/hip_bf16.h>
#include <stdint.h>
#include <stddef.h>

// ---------------------------------------------------------------------------
// Sliding-window attention, MI455X (gfx1250, wave32, WMMA 16x16x32 bf16).
// Grid: B * (S/64) blocks; block = 256 threads = 8 waves; 1 block per WGP
// (LDS-bound: 267 KB of the 320 KB WGP LDS).
// ---------------------------------------------------------------------------

#define B_      4
#define S_      4096
#define E_      1024
#define W_      256
#define BQ      64               // query rows per block
#define KBAND   576              // BQ + 2*W  (band of keys for this block)
#define CHUNK   32               // K/V rows staged per iteration
#define NCHUNK  18               // KBAND / CHUNK
#define QS_STRIDE 1032           // E_ + 8 bf16 pad   (2064 B, 16B-aligned rows)
#define KS_STRIDE 1032
#define PS_STRIDE 584            // KBAND + 8         (1168 B rows)
#define VT_STRIDE 40             // CHUNK + 8         (80 B rows, bank-spread)

typedef __attribute__((ext_vector_type(4)))  __bf16        v4bf;
typedef __attribute__((ext_vector_type(16))) __bf16        v16bf;
typedef __attribute__((ext_vector_type(8)))  float         v8f;
typedef __attribute__((ext_vector_type(8)))  unsigned short us8;
typedef __attribute__((ext_vector_type(16))) unsigned short us16;

// LDS layout (bytes). Vt (phase 2) aliases the Q buffer (dead by then).
#define OFF_QV  0
#define SZ_QV   (BQ * QS_STRIDE * 2)        // 132096 (Vt needs 1024*40*2=81920)
#define OFF_K   (OFF_QV + SZ_QV)
#define SZ_K    (CHUNK * KS_STRIDE * 2)     // 66048
#define OFF_P   (OFF_K + SZ_K)
#define SZ_P    (BQ * PS_STRIDE * 2)        // 74752
#define OFF_MAX (OFF_P + SZ_P)
#define OFF_SUM (OFF_MAX + BQ * 4)
#define LDS_BYTES (OFF_SUM + BQ * 4)        // 273408 <= 327680

// Load 16 bf16 (two 16-byte groups) into a v16bf fragment via 2x ds_load_b128.
static __device__ __forceinline__ v16bf ld_pair(const __bf16* p0, const __bf16* p1) {
  us8 lo = *(const us8*)p0;
  us8 hi = *(const us8*)p1;
  us16 u = __builtin_shufflevector(lo, hi, 0,1,2,3,4,5,6,7,8,9,10,11,12,13,14,15);
  return __builtin_bit_cast(v16bf, u);
}

__global__ __launch_bounds__(256, 1)
void SlidingWindowAttention_6854767804603_kernel(const float* __restrict__ Qg,
                                                 const float* __restrict__ Kg,
                                                 const float* __restrict__ Vg,
                                                 const unsigned char* __restrict__ Mg,
                                                 float* __restrict__ Og) {
  __shared__ __align__(16) unsigned char lds[LDS_BYTES];
  __bf16* Qs = (__bf16*)(lds + OFF_QV);     // [64][1032] bf16, phase 0/1
  __bf16* Vt = (__bf16*)(lds + OFF_QV);     // [1024][40]  bf16, phase 2 (alias)
  __bf16* Ks = (__bf16*)(lds + OFF_K);      // [32][1032]  bf16
  __bf16* Ps = (__bf16*)(lds + OFF_P);      // [64][584]   bf16 softmax probs
  float*  rowmax = (float*)(lds + OFF_MAX); // [64]
  float*  rowsum = (float*)(lds + OFF_SUM); // [64]

  const int t    = threadIdx.x;
  const int lane = t & 31;
  const int wave = t >> 5;
  const int qsub = wave & 3;                // which 16-row query subtile
  const int ksel = wave >> 2;               // which 16-key half of the 32-chunk
  const int m15  = lane & 15;
  const int hi16 = (lane >> 4) & 1;

  const int nqb = S_ / BQ;
  const int b   = blockIdx.x / nqb;
  const int qb0 = (blockIdx.x % nqb) * BQ;
  const int kb0 = qb0 - W_;                 // band start (may be negative)

  if (t < BQ) { rowmax[t] = -3.0e38f; rowsum[t] = 0.0f; }

  // ---- Phase 0: stage Q block (64x1024 fp32 -> bf16 LDS, row-major) ----
  {
    const float* qp = Qg + ((size_t)b * S_ + qb0) * E_ + t * 4;
    #pragma unroll 4
    for (int r = 0; r < BQ; ++r) {
      float4 v = *(const float4*)(qp + (size_t)r * E_);
      v4bf o = { (__bf16)v.x, (__bf16)v.y, (__bf16)v.z, (__bf16)v.w };
      *(v4bf*)(Qs + r * QS_STRIDE + t * 4) = o;
    }
  }

  // ---- Phase 1: S = Q K^T over the band; 18 register-resident score tiles ----
  v8f acc[NCHUNK] = {};

  #pragma unroll
  for (int c = 0; c < NCHUNK; ++c) {
    __syncthreads();
    // cooperative stage of K chunk c (32 rows), fp32 -> bf16
    {
      #pragma unroll 4
      for (int r = 0; r < CHUNK; ++r) {
        int krow = kb0 + c * CHUNK + r;
        float4 v = {0.f, 0.f, 0.f, 0.f};
        if (krow >= 0 && krow < S_)
          v = *(const float4*)(Kg + ((size_t)b * S_ + krow) * E_ + t * 4);
        v4bf o = { (__bf16)v.x, (__bf16)v.y, (__bf16)v.z, (__bf16)v.w };
        *(v4bf*)(Ks + r * KS_STRIDE + t * 4) = o;
      }
      if (c + 1 < NCHUNK) {
        int krow = kb0 + (c + 1) * CHUNK + (t >> 3);
        if (krow >= 0 && krow < S_)
          __builtin_prefetch(Kg + ((size_t)b * S_ + krow) * E_ + (t & 7) * 128, 0, 0);
      }
    }
    __syncthreads();
    // this wave's 16x16 score tile: rows qsub*16.., keys ksel*16.. of the chunk
    const __bf16* qbase = Qs + (qsub * 16 + m15) * QS_STRIDE + hi16 * 8;   // A frag
    const __bf16* kbase = Ks + (ksel * 16 + m15) * KS_STRIDE + hi16 * 8;   // B frag
    for (int e0 = 0; e0 < E_; e0 += 32) {
      v16bf a  = ld_pair(qbase + e0, qbase + e0 + 16);
      v16bf bb = ld_pair(kbase + e0, kbase + e0 + 16);
      acc[c] = __builtin_amdgcn_wmma_f32_16x16x32_bf16(
                   false, a, false, bb, (short)0, acc[c], false, false);
    }
  }

  // ---- Softmax over the 576-wide band (register scores + LDS float atomics) ----
  const float scale = 0.03125f;             // 1/sqrt(1024)
  const int rowoff = qsub * 16 + hi16 * 8;  // C-frag: VGPR v -> row rowoff+v, col m15
  float rmax[8];
  #pragma unroll
  for (int v = 0; v < 8; ++v) rmax[v] = -3.0e38f;

  #pragma unroll
  for (int c = 0; c < NCHUNK; ++c) {
    int k = kb0 + (2 * c + ksel) * 16 + m15;
    bool kin = (k >= 0) && (k < S_);
    if (kin) kin = (Mg[(size_t)b * S_ + k] != 0);
    #pragma unroll
    for (int v = 0; v < 8; ++v) {
      int q = qb0 + rowoff + v;
      float sc = acc[c][v] * scale;
      int d = q - k; if (d < 0) d = -d;
      sc = (kin && d <= W_) ? sc : -1.0e9f;
      acc[c][v] = sc;
      rmax[v] = fmaxf(rmax[v], sc);
    }
  }
  #pragma unroll
  for (int v = 0; v < 8; ++v)
    __hip_atomic_fetch_max(&rowmax[rowoff + v], rmax[v],
                           __ATOMIC_RELAXED, __HIP_MEMORY_SCOPE_WORKGROUP);
  __syncthreads();

  float rmx[8], rsum[8];
  #pragma unroll
  for (int v = 0; v < 8; ++v) { rmx[v] = rowmax[rowoff + v]; rsum[v] = 0.f; }
  #pragma unroll
  for (int c = 0; c < NCHUNK; ++c) {
    #pragma unroll
    for (int v = 0; v < 8; ++v)
      rsum[v] += exp2f((acc[c][v] - rmx[v]) * 1.4426950408889634f);
  }
  #pragma unroll
  for (int v = 0; v < 8; ++v)
    __hip_atomic_fetch_add(&rowsum[rowoff + v], rsum[v],
                           __ATOMIC_RELAXED, __HIP_MEMORY_SCOPE_WORKGROUP);
  __syncthreads();

  float rinv[8];
  #pragma unroll
  for (int v = 0; v < 8; ++v) {
    float s = rowsum[rowoff + v];
    rinv[v] = (s > 0.f) ? (1.0f / s) : 0.f;
  }
  #pragma unroll
  for (int c = 0; c < NCHUNK; ++c) {
    int kc = (2 * c + ksel) * 16 + m15;
    #pragma unroll
    for (int v = 0; v < 8; ++v) {
      float p = exp2f((acc[c][v] - rmx[v]) * 1.4426950408889634f) * rinv[v];
      Ps[(rowoff + v) * PS_STRIDE + kc] = (__bf16)p;
    }
  }

  // ---- Phase 2: O = P V; V staged transposed; 16 out tiles/wave, 2 passes ----
  float* ob = Og + ((size_t)b * S_ + qb0 + rowoff) * E_;
  #pragma unroll
  for (int pass = 0; pass < 2; ++pass) {
    v8f oacc[16] = {};
    for (int c = 0; c < NCHUNK; ++c) {
      __syncthreads();
      // stage V chunk c transposed: Vt[e][k] = bf16(V[kb0+c*32+k][e])
      #pragma unroll 4
      for (int r = 0; r < CHUNK; ++r) {
        int krow = kb0 + c * CHUNK + r;
        float4 v = {0.f, 0.f, 0.f, 0.f};
        if (krow >= 0 && krow < S_)
          v = *(const float4*)(Vg + ((size_t)b * S_ + krow) * E_ + t * 4);
        Vt[(t * 4 + 0) * VT_STRIDE + r] = (__bf16)v.x;
        Vt[(t * 4 + 1) * VT_STRIDE + r] = (__bf16)v.y;
        Vt[(t * 4 + 2) * VT_STRIDE + r] = (__bf16)v.z;
        Vt[(t * 4 + 3) * VT_STRIDE + r] = (__bf16)v.w;
      }
      __syncthreads();
      const __bf16* pb = Ps + (qsub * 16 + m15) * PS_STRIDE + c * 32 + hi16 * 8;
      v16bf a = ld_pair(pb, pb + 16);          // A = P tile (shared across j)
      #pragma unroll
      for (int j = 0; j < 16; ++j) {
        int e0 = (pass * 32 + 2 * j + ksel) * 16;
        const __bf16* vb = Vt + (e0 + m15) * VT_STRIDE + hi16 * 8;
        v16bf bv = ld_pair(vb, vb + 16);       // B = V columns (contiguous in Vt)
        oacc[j] = __builtin_amdgcn_wmma_f32_16x16x32_bf16(
                      false, a, false, bv, (short)0, oacc[j], false, false);
      }
    }
    #pragma unroll
    for (int j = 0; j < 16; ++j) {
      int e0 = (pass * 32 + 2 * j + ksel) * 16;
      #pragma unroll
      for (int v = 0; v < 8; ++v)
        ob[(size_t)v * E_ + e0 + m15] = oacc[j][v];
    }
  }
}

extern "C" void kernel_launch(void* const* d_in, const int* in_sizes, int n_in,
                              void* d_out, int out_size, void* d_ws, size_t ws_size,
                              hipStream_t stream) {
  (void)in_sizes; (void)n_in; (void)out_size; (void)d_ws; (void)ws_size;
  const float* Q = (const float*)d_in[0];
  const float* K = (const float*)d_in[1];
  const float* V = (const float*)d_in[2];
  const unsigned char* M = (const unsigned char*)d_in[3];
  float* O = (float*)d_out;
  dim3 grid(B_ * (S_ / BQ));   // 256 blocks
  dim3 block(256);             // 8 wave32 waves
  hipLaunchKernelGGL(SlidingWindowAttention_6854767804603_kernel,
                     grid, block, 0, stream, Q, K, V, M, O);
}
